// PointNet2Encoder_32452772889069
// MI455X (gfx1250) — matrix operand order repaired and checked by reading, not compiled, not run on hardware
//
#include <hip/hip_runtime.h>

typedef __attribute__((ext_vector_type(16))) _Float16 v16h_t;
typedef __attribute__((ext_vector_type(8)))  float    v8f_t;

// ---------------------------------------------------------------------------
// Problem constants
// ---------------------------------------------------------------------------
#define BATCH      8          // 4*2 flattened
#define NPTS       8192
#define CIN_PC     9
#define NPOINT1    1024
#define NPOINT2    256
#define F1_C       96
#define F2_C       256
#define C_OUT      128

// Workspace byte offsets (fixed layout; all 32B aligned)
#define WS_FPS1    ((size_t)0)          // [8*1024] int
#define WS_NXYZ1   ((size_t)32768)      // [8*1024*3] f32
#define WS_BALL1A  ((size_t)131072)     // [8*1024*16] int
#define WS_BALL1B  ((size_t)655360)     // [8*1024*32] int
#define WS_F1      ((size_t)1703936)    // [8*1024*96] f16
#define WS_FPS2    ((size_t)4849664)    // [8*256] int
#define WS_NXYZ2   ((size_t)4857856)    // [8*256*3] f32
#define WS_BALL2A  ((size_t)4882432)    // [8*256*16] int
#define WS_BALL2B  ((size_t)5013504)    // [8*256*32] int
#define WS_F2      ((size_t)5275648)    // [8*256*256] f16
#define WS_WEIGHTS ((size_t)7372800)    // fp16 fragment-packed weights

// ---------------------------------------------------------------------------
// Farthest point sampling: one block per batch, sequential scan with
// block-wide argmax reduction, LDS-resident running min-distance.
// ---------------------------------------------------------------------------
template <int N, int NPOINT>
__global__ void __launch_bounds__(256)
fps_kernel(const float* __restrict__ xyz, int stride, int* __restrict__ out) {
  const int b = blockIdx.x;
  const float* base = xyz + (size_t)b * N * stride;
  __shared__ float dist[N];
  __shared__ float red[256];
  __shared__ int   redi[256];
  __shared__ int   s_far;
  const int t = threadIdx.x;
  for (int i = t; i < N; i += 256) dist[i] = 1e10f;
  if (t == 0) s_far = 0;
  __syncthreads();
  for (int it = 0; it < NPOINT; ++it) {
    const int far = s_far;
    if (t == 0) out[b * NPOINT + it] = far;
    const float cx = base[far * stride + 0];
    const float cy = base[far * stride + 1];
    const float cz = base[far * stride + 2];
    float bestd = -1.0f;
    int   besti = 0;
    for (int i = t; i < N; i += 256) {
      const float dx = base[i * stride + 0] - cx;
      const float dy = base[i * stride + 1] - cy;
      const float dz = base[i * stride + 2] - cz;
      const float d  = dx * dx + dy * dy + dz * dz;
      const float nd = fminf(dist[i], d);
      dist[i] = nd;
      if (nd > bestd) { bestd = nd; besti = i; }
    }
    red[t] = bestd; redi[t] = besti;
    __syncthreads();
    for (int s = 128; s > 0; s >>= 1) {
      if (t < s && red[t + s] > red[t]) { red[t] = red[t + s]; redi[t] = redi[t + s]; }
      __syncthreads();
    }
    if (t == 0) s_far = redi[0];
    __syncthreads();
  }
}

// ---------------------------------------------------------------------------
// Gather sampled centroid coordinates.
// ---------------------------------------------------------------------------
__global__ void gather_xyz_kernel(const float* __restrict__ xyz, int stride, int N,
                                  const int* __restrict__ idx, int npoint,
                                  float* __restrict__ out, int total) {
  const int i = blockIdx.x * blockDim.x + threadIdx.x;
  if (i >= total) return;
  const int b = i / npoint;
  const int p = idx[i];
  const float* s = xyz + ((size_t)b * N + p) * stride;
  out[(size_t)i * 3 + 0] = s[0];
  out[(size_t)i * 3 + 1] = s[1];
  out[(size_t)i * 3 + 2] = s[2];
}

// ---------------------------------------------------------------------------
// Ball query: first-ns-within-radius semantics, pad with first hit (or 0).
// ---------------------------------------------------------------------------
__global__ void ball_query_kernel(const float* __restrict__ xyz, int stride, int N,
                                  const float* __restrict__ new_xyz, int npoint,
                                  float r2, int ns, int* __restrict__ out, int total) {
  const int i = blockIdx.x * blockDim.x + threadIdx.x;
  if (i >= total) return;
  const int b = i / npoint;
  const float* base = xyz + (size_t)b * N * stride;
  const float qx = new_xyz[(size_t)i * 3 + 0];
  const float qy = new_xyz[(size_t)i * 3 + 1];
  const float qz = new_xyz[(size_t)i * 3 + 2];
  int* o = out + (size_t)i * ns;
  int cnt = 0;
  for (int n = 0; n < N && cnt < ns; ++n) {
    const float dx = base[n * stride + 0] - qx;
    const float dy = base[n * stride + 1] - qy;
    const float dz = base[n * stride + 2] - qz;
    if (dx * dx + dy * dy + dz * dz < r2) o[cnt++] = n;
  }
  const int first = (cnt > 0) ? o[0] : 0;
  for (int k = cnt; k < ns; ++k) o[k] = first;
}

// ---------------------------------------------------------------------------
// Weight prep: f32 [K, C] -> fp16 fragment-packed, zero-padded to
// [Kpad/32 tiles][Cpad/16 tiles][32 lanes][16 halfs], so a wave's B fragment
// for tile (kt,nt) is one aligned 32-byte vector load per lane.
// B layout: lane<16 -> col = nt*16+lane, K = kt*32 + {0..7,16..23};
//           lane>=16 -> col = nt*16+lane-16, K = kt*32 + {8..15,24..31}.
// ---------------------------------------------------------------------------
__global__ void prep_weight_frag_kernel(const float* __restrict__ W, int K, int C,
                                        int NT, int total, _Float16* __restrict__ out) {
  const int i = blockIdx.x * blockDim.x + threadIdx.x;
  if (i >= total) return;
  const int e    = i & 15;
  const int lane = (i >> 4) & 31;
  const int tile = i >> 9;
  const int nt = tile % NT;
  const int kt = tile / NT;
  const int k = kt * 32 + ((lane < 16) ? 0 : 8) + (e & 7) + ((e < 8) ? 0 : 16);
  const int c = nt * 16 + (lane & 15);
  const float v = (k < K && c < C) ? W[k * C + c] : 0.0f;
  out[i] = (_Float16)v;
}

// ---------------------------------------------------------------------------
// One fully-unrolled WMMA MLP layer on a wave-private LDS activation tile.
// A fragments hoisted into registers once per M tile; B fragments are single
// 32-byte vector loads from the fragment-packed weight buffer.
// A layout (16-bit A 16x32): lane<16 -> row=lane, K={0..7,16..23};
// lane>=16 -> row=lane-16, K={8..15,24..31}.
// C/D: VGPR r, lane<16 -> (M=r, N=lane); lane>=16 -> (M=8+r, N=lane-16).
// ---------------------------------------------------------------------------
template <int KPAD, int COUT, int MT>
__device__ inline void wmma_layer(const _Float16* __restrict__ inA,
                                  _Float16* __restrict__ outB,
                                  const v16h_t* __restrict__ Wfrag,
                                  const float* __restrict__ bias, int lane) {
  constexpr int KT = KPAD / 32;
  constexpr int NT = COUT / 16;
  const int l15 = lane & 15;
  const bool lo = lane < 16;
#pragma unroll
  for (int mt = 0; mt < MT; ++mt) {
    v16h_t af[KT];
    const int mrow = mt * 16 + l15;
#pragma unroll
    for (int kt = 0; kt < KT; ++kt) {
      const int ka = kt * 32 + (lo ? 0 : 8);
#pragma unroll
      for (int e = 0; e < 8; ++e) {
        af[kt][e]     = inA[mrow * 128 + ka + e];
        af[kt][e + 8] = inA[mrow * 128 + ka + 16 + e];
      }
    }
#pragma unroll
    for (int nt = 0; nt < NT; ++nt) {
      v8f_t acc = {0.f, 0.f, 0.f, 0.f, 0.f, 0.f, 0.f, 0.f};
#pragma unroll
      for (int kt = 0; kt < KT; ++kt) {
        const v16h_t bf = Wfrag[(kt * NT + nt) * 32 + lane];
        acc = __builtin_amdgcn_wmma_f32_16x16x32_f16(false, af[kt], false, bf,
                                                     (short)0, acc, false, false);
      }
      const int ncol = nt * 16 + l15;
      const float bv = bias[ncol];
      const int rbase = mt * 16 + (lo ? 0 : 8);
#pragma unroll
      for (int r = 0; r < 8; ++r) {
        const float v = acc[r] + bv;
        outB[(rbase + r) * 128 + ncol] = (_Float16)(v > 0.f ? v : 0.f);
      }
    }
  }
}

// ---------------------------------------------------------------------------
// Grouped MSG set-abstraction MLP: one wave per centroid group.
// Fully compile-time layer configuration; fp16 activations throughout.
// ---------------------------------------------------------------------------
template <typename FT, int NS, int CFEAT,
          int K0, int C0, int K1, int C1, int K2, int C2>
__global__ void __launch_bounds__(64)
sa_mlp_kernel(const float* __restrict__ src_xyz, int xyz_stride,
              const FT* __restrict__ src_feat, int feat_stride,
              const float* __restrict__ new_xyz,
              const int* __restrict__ ball, int npoint, int Nsrc,
              const _Float16* __restrict__ W0, const float* __restrict__ b0,
              const _Float16* __restrict__ W1, const float* __restrict__ b1,
              const _Float16* __restrict__ W2, const float* __restrict__ b2,
              _Float16* __restrict__ fout, int fout_stride, int col_off) {
  const int lane = threadIdx.x & 31;
  const int wave = threadIdx.x >> 5;
  const int g = blockIdx.x * 2 + wave;      // grid sized exactly
  const int b = g / npoint;

  __builtin_prefetch((const void*)W0, 0, 1);
  __builtin_prefetch((const void*)W1, 0, 1);
  __builtin_prefetch((const void*)W2, 0, 1);

  __shared__ __align__(32) _Float16 act[2][2][32 * 128];
  _Float16* A  = &act[wave][0][0];
  _Float16* Bf = &act[wave][1][0];

  // zero both ping-pong buffers (guarantees K padding through all layers)
  for (int i = lane; i < 32 * 128; i += 32) { A[i] = (_Float16)0.f; Bf[i] = (_Float16)0.f; }
  __syncthreads();

  // stage grouped inputs: row = neighbor sample, cols = [xyz diff | feats]
  const float* bx = src_xyz  + (size_t)b * Nsrc * xyz_stride;
  const FT*    bf = src_feat + (size_t)b * Nsrc * feat_stride;
  const float qx = new_xyz[(size_t)g * 3 + 0];
  const float qy = new_xyz[(size_t)g * 3 + 1];
  const float qz = new_xyz[(size_t)g * 3 + 2];
  if (lane < NS) {
    const int p = ball[(size_t)g * NS + lane];
    const float* px = bx + (size_t)p * xyz_stride;
    A[lane * 128 + 0] = (_Float16)(px[0] - qx);
    A[lane * 128 + 1] = (_Float16)(px[1] - qy);
    A[lane * 128 + 2] = (_Float16)(px[2] - qz);
    const FT* pf = bf + (size_t)p * feat_stride;
#pragma unroll 8
    for (int c = 0; c < CFEAT; ++c) A[lane * 128 + 3 + c] = (_Float16)pf[c];
  }
  __syncthreads();

  constexpr int MT = NS / 16;
  wmma_layer<K0, C0, MT>(A,  Bf, (const v16h_t*)W0, b0, lane);
  __syncthreads();
  wmma_layer<K1, C1, MT>(Bf, A,  (const v16h_t*)W1, b1, lane);
  __syncthreads();
  wmma_layer<K2, C2, MT>(A,  Bf, (const v16h_t*)W2, b2, lane);
  __syncthreads();

  // max-pool over neighbors (ReLU output => >= 0)
#pragma unroll
  for (int c = lane; c < C2; c += 32) {
    float m = 0.f;
#pragma unroll
    for (int s = 0; s < NS; ++s) m = fmaxf(m, (float)Bf[s * 128 + c]);
    fout[(size_t)g * fout_stride + col_off + c] = (_Float16)m;
  }
}

// ---------------------------------------------------------------------------
// Final linear: out[:, 3:128] = f2 @ Wlin + b, out[:, 0:3] = new_xyz2.
// One wave per 16-row tile; fp16 A tile read straight from global into
// register fragments (no LDS), 8 N-tiles x 8 K-tiles of WMMA.
// ---------------------------------------------------------------------------
__global__ void __launch_bounds__(64)
final_linear_kernel(const _Float16* __restrict__ f2,   // [2048 x 256] fp16
                    const v16h_t* __restrict__ Wfrag,  // frag-packed [256 x 128]
                    const float* __restrict__ blin,    // [125]
                    const float* __restrict__ nxyz2,   // [2048 x 3]
                    float* __restrict__ out) {         // [2048 x 128]
  const int lane = threadIdx.x & 31;
  const int wave = threadIdx.x >> 5;
  const int tile = blockIdx.x * 2 + wave;              // 128 tiles exactly
  const int row0 = tile * 16;
  const int l15 = lane & 15;
  const bool lo = lane < 16;

  __builtin_prefetch((const void*)Wfrag, 0, 1);

  const _Float16* Arow = f2 + ((size_t)row0 + l15) * 256;
  v16h_t af[8];
#pragma unroll
  for (int kt = 0; kt < 8; ++kt) {
    const int ka = kt * 32 + (lo ? 0 : 8);
#pragma unroll
    for (int e = 0; e < 8; ++e) {
      af[kt][e]     = Arow[ka + e];
      af[kt][e + 8] = Arow[ka + 16 + e];
    }
  }

#pragma unroll
  for (int nt = 0; nt < 8; ++nt) {
    v8f_t acc = {0.f, 0.f, 0.f, 0.f, 0.f, 0.f, 0.f, 0.f};
#pragma unroll
    for (int kt = 0; kt < 8; ++kt) {
      const v16h_t bf = Wfrag[(kt * 8 + nt) * 32 + lane];
      acc = __builtin_amdgcn_wmma_f32_16x16x32_f16(false, af[kt], false, bf,
                                                   (short)0, acc, false, false);
    }
    const int ncol = nt * 16 + l15;
    if (ncol < 125) {
      const float bv = blin[ncol];
      const int rbase = row0 + (lo ? 0 : 8);
#pragma unroll
      for (int r = 0; r < 8; ++r)
        out[(size_t)(rbase + r) * C_OUT + 3 + ncol] = acc[r] + bv;
    }
  }
  if (lane < 16) {
    const int row = row0 + lane;
    out[(size_t)row * C_OUT + 0] = nxyz2[(size_t)row * 3 + 0];
    out[(size_t)row * C_OUT + 1] = nxyz2[(size_t)row * 3 + 1];
    out[(size_t)row * C_OUT + 2] = nxyz2[(size_t)row * 3 + 2];
  }
}

// ---------------------------------------------------------------------------
// Host orchestration
// ---------------------------------------------------------------------------
static inline int round_up(int v, int m) { return (v + m - 1) / m * m; }

extern "C" void kernel_launch(void* const* d_in, const int* in_sizes, int n_in,
                              void* d_out, int out_size, void* d_ws, size_t ws_size,
                              hipStream_t stream) {
  (void)in_sizes; (void)n_in; (void)out_size; (void)ws_size;

  const float* pc = (const float*)d_in[0];     // [8, 8192, 9]
  char* ws = (char*)d_ws;

  int*      fps1   = (int*)     (ws + WS_FPS1);
  float*    nxyz1  = (float*)   (ws + WS_NXYZ1);
  int*      ball1a = (int*)     (ws + WS_BALL1A);
  int*      ball1b = (int*)     (ws + WS_BALL1B);
  _Float16* f1     = (_Float16*)(ws + WS_F1);
  int*      fps2   = (int*)     (ws + WS_FPS2);
  float*    nxyz2  = (float*)   (ws + WS_NXYZ2);
  int*      ball2a = (int*)     (ws + WS_BALL2A);
  int*      ball2b = (int*)     (ws + WS_BALL2B);
  _Float16* f2     = (_Float16*)(ws + WS_F2);
  _Float16* wbase  = (_Float16*)(ws + WS_WEIGHTS);

  // Weight specs: {d_in index, K, Cout}; flattening order of setup_inputs dict.
  struct WS_ { int idx, K, C; } wspec[13] = {
    {1, 9, 16},  {3, 16, 16},  {5, 16, 32},     // sa1 mlp0
    {7, 9, 32},  {9, 32, 32},  {11, 32, 64},    // sa1 mlp1
    {13, 99, 64},{15, 64, 64}, {17, 64, 128},   // sa2 mlp0
    {19, 99, 64},{21, 64, 96}, {23, 96, 128},   // sa2 mlp1
    {25, 256, 125}                              // final linear
  };
  _Float16* wptr[13];
  size_t off = 0;
  for (int i = 0; i < 13; ++i) {
    const int Kpad = round_up(wspec[i].K, 32);
    const int Cpad = round_up(wspec[i].C, 16);
    wptr[i] = wbase + off;
    const int total = Kpad * Cpad;
    prep_weight_frag_kernel<<<(total + 255) / 256, 256, 0, stream>>>(
        (const float*)d_in[wspec[i].idx], wspec[i].K, wspec[i].C,
        Cpad / 16, total, wptr[i]);
    off += (size_t)total;
  }

  // ---- SA1 ----
  fps_kernel<NPTS, NPOINT1><<<BATCH, 256, 0, stream>>>(pc, CIN_PC, fps1);

  {
    const int total = BATCH * NPOINT1;
    gather_xyz_kernel<<<(total + 255) / 256, 256, 0, stream>>>(
        pc, CIN_PC, NPTS, fps1, NPOINT1, nxyz1, total);
    ball_query_kernel<<<(total + 255) / 256, 256, 0, stream>>>(
        pc, CIN_PC, NPTS, nxyz1, NPOINT1, 0.05f * 0.05f, 16, ball1a, total);
    ball_query_kernel<<<(total + 255) / 256, 256, 0, stream>>>(
        pc, CIN_PC, NPTS, nxyz1, NPOINT1, 0.10f * 0.10f, 32, ball1b, total);

    sa_mlp_kernel<float, 16, 6, 32, 16, 32, 16, 32, 32>
        <<<total / 2, 64, 0, stream>>>(
        pc, CIN_PC, pc + 3, CIN_PC, nxyz1, ball1a, NPOINT1, NPTS,
        wptr[0], (const float*)d_in[2],
        wptr[1], (const float*)d_in[4],
        wptr[2], (const float*)d_in[6],
        f1, F1_C, 0);

    sa_mlp_kernel<float, 32, 6, 32, 32, 32, 32, 32, 64>
        <<<total / 2, 64, 0, stream>>>(
        pc, CIN_PC, pc + 3, CIN_PC, nxyz1, ball1b, NPOINT1, NPTS,
        wptr[3], (const float*)d_in[8],
        wptr[4], (const float*)d_in[10],
        wptr[5], (const float*)d_in[12],
        f1, F1_C, 32);
  }

  // ---- SA2 ----
  fps_kernel<NPOINT1, NPOINT2><<<BATCH, 256, 0, stream>>>(nxyz1, 3, fps2);

  {
    const int total = BATCH * NPOINT2;
    gather_xyz_kernel<<<(total + 255) / 256, 256, 0, stream>>>(
        nxyz1, 3, NPOINT1, fps2, NPOINT2, nxyz2, total);
    ball_query_kernel<<<(total + 255) / 256, 256, 0, stream>>>(
        nxyz1, 3, NPOINT1, nxyz2, NPOINT2, 0.10f * 0.10f, 16, ball2a, total);
    ball_query_kernel<<<(total + 255) / 256, 256, 0, stream>>>(
        nxyz1, 3, NPOINT1, nxyz2, NPOINT2, 0.20f * 0.20f, 32, ball2b, total);

    sa_mlp_kernel<_Float16, 16, 96, 128, 64, 64, 64, 64, 128>
        <<<total / 2, 64, 0, stream>>>(
        nxyz1, 3, f1, F1_C, nxyz2, ball2a, NPOINT2, NPOINT1,
        wptr[6], (const float*)d_in[14],
        wptr[7], (const float*)d_in[16],
        wptr[8], (const float*)d_in[18],
        f2, F2_C, 0);

    sa_mlp_kernel<_Float16, 32, 96, 128, 64, 64, 96, 96, 128>
        <<<total / 2, 64, 0, stream>>>(
        nxyz1, 3, f1, F1_C, nxyz2, ball2b, NPOINT2, NPOINT1,
        wptr[9],  (const float*)d_in[20],
        wptr[10], (const float*)d_in[22],
        wptr[11], (const float*)d_in[24],
        f2, F2_C, 128);
  }

  // ---- final linear + xyz concat ----
  final_linear_kernel<<<64, 64, 0, stream>>>(
      f2, (const v16h_t*)wptr[12], (const float*)d_in[26], nxyz2, (float*)d_out);
}
